// Qwen3Attention_71116068487419
// MI455X (gfx1250) — matrix operand order repaired and checked by reading, not compiled, not run on hardware
//
#include <hip/hip_runtime.h>

// ---------------------------------------------------------------------------
// Qwen3-style gated attention block for gfx1250 (CDNA5, wave32, WMMA).
// All matmuls run on v_wmma_f32_16x16x32_bf16 (bf16 in / f32 accumulate).
// Flash attention prefetches K tiles with global_load_async_to_lds_b128
// (ASYNCcnt) into double-buffered LDS; GEMM is software-pipelined with
// double-buffered LDS (one barrier per K step).
// ---------------------------------------------------------------------------

typedef __attribute__((ext_vector_type(16))) __bf16 v16bf;
typedef __attribute__((ext_vector_type(8)))  __bf16 v8bf;
typedef __attribute__((ext_vector_type(4)))  __bf16 v4bf;
typedef __attribute__((ext_vector_type(8)))  float  v8f;

#define S_LEN 2048
#define HSZ   4096
#define NH    32
#define NKV   8
#define DH    128
#define SOFT_SCALE 0.08838834764831845f   // 1/sqrt(128)

union Frag { v16bf v; v8bf h8[2]; };

__device__ __forceinline__ v8f wmma_bf16(v16bf a, v16bf b, v8f c) {
  return __builtin_amdgcn_wmma_f32_16x16x32_bf16(false, a, false, b, (short)0, c,
                                                 false, false);
}

// ---------------------------------------------------------------------------
// Generic tiled GEMM: C[M,N](f32) = A[M,K] x B[K,N](f32 weights)
// A_BF16 is a compile-time flag (A is bf16 vs f32-converted-to-bf16).
// Block tile 128x128, K-step 32, 8 waves each computing 64x32 (4x2 WMMA tiles).
// Register prefetch + double-buffered LDS: one barrier per K step.
// ---------------------------------------------------------------------------
template <int A_BF16>
__global__ __launch_bounds__(256) void gemm_wmma_kernel(
    const void* __restrict__ Aptr, const float* __restrict__ Bw,
    float* __restrict__ C, int N, int K)
{
  __shared__ __align__(16) __bf16 Asm[2][128 * 40];  // [m][k], stride 40
  __shared__ __align__(16) __bf16 Bsm[2][128 * 40];  // [n][k], stride 40

  const int tid  = threadIdx.x;
  const int wave = tid >> 5, lane = tid & 31;
  const int hl   = lane >> 4, mr = lane & 15;
  const int wm   = wave & 1, wn = wave >> 1;        // 2 waves (M) x 4 waves (N)
  const int m0   = blockIdx.y * 128, n0 = blockIdx.x * 128;

  const float*  Af = (const float*)Aptr;
  const __bf16* Ab = (const __bf16*)Aptr;

  // Per-thread staging coordinates (constant over K iterations).
  int arow[4], ac4[4], bkr[4], bc4[4];
#pragma unroll
  for (int l = 0; l < 4; l++) {
    int e = l * 256 + tid;               // 0..1023 quads
    arow[l] = e >> 3; ac4[l] = e & 7;    // A: 8 quads per 32-wide row
    bkr[l]  = e >> 5; bc4[l] = e & 31;   // B: 32 quads per 128-wide k-row
  }

  const v8f zf = {0.f,0.f,0.f,0.f,0.f,0.f,0.f,0.f};
  v8f acc[4][2];
#pragma unroll
  for (int i = 0; i < 4; i++)
#pragma unroll
    for (int j = 0; j < 2; j++) acc[i][j] = zf;

  float4 aF[4]; v4bf aH[4]; float4 bF[4];

  auto load_global = [&](int k0) {
#pragma unroll
    for (int l = 0; l < 4; l++) {
      if (A_BF16)
        aH[l] = *(const v4bf*)(Ab + (size_t)(m0 + arow[l]) * K + k0 + ac4[l] * 4);
      else
        aF[l] = *(const float4*)(Af + (size_t)(m0 + arow[l]) * K + k0 + ac4[l] * 4);
    }
#pragma unroll
    for (int l = 0; l < 4; l++)
      bF[l] = *(const float4*)(Bw + (size_t)(k0 + bkr[l]) * N + n0 + bc4[l] * 4);
  };

  auto stage_lds = [&](int buf) {
#pragma unroll
    for (int l = 0; l < 4; l++) {
      v4bf st;
      if (A_BF16) {
        st = aH[l];
      } else {
        st[0] = (__bf16)aF[l].x; st[1] = (__bf16)aF[l].y;
        st[2] = (__bf16)aF[l].z; st[3] = (__bf16)aF[l].w;
      }
      *(v4bf*)&Asm[buf][arow[l] * 40 + ac4[l] * 4] = st;
    }
#pragma unroll
    for (int l = 0; l < 4; l++) {
      Bsm[buf][(bc4[l] * 4 + 0) * 40 + bkr[l]] = (__bf16)bF[l].x;
      Bsm[buf][(bc4[l] * 4 + 1) * 40 + bkr[l]] = (__bf16)bF[l].y;
      Bsm[buf][(bc4[l] * 4 + 2) * 40 + bkr[l]] = (__bf16)bF[l].z;
      Bsm[buf][(bc4[l] * 4 + 3) * 40 + bkr[l]] = (__bf16)bF[l].w;
    }
  };

  load_global(0);
  int buf = 0;
  for (int k0 = 0; k0 < K; k0 += 32, buf ^= 1) {
    stage_lds(buf);
    __syncthreads();
    if (k0 + 32 < K) load_global(k0 + 32);   // overlap with WMMAs below

    Frag af[4], bf_[2];
#pragma unroll
    for (int i = 0; i < 4; i++) {
      const __bf16* ap = &Asm[buf][(wm * 64 + i * 16 + mr) * 40];
      af[i].h8[0] = *(const v8bf*)(ap + 8 * hl);
      af[i].h8[1] = *(const v8bf*)(ap + 16 + 8 * hl);
    }
#pragma unroll
    for (int j = 0; j < 2; j++) {
      const __bf16* bp = &Bsm[buf][(wn * 32 + j * 16 + mr) * 40];
      bf_[j].h8[0] = *(const v8bf*)(bp + 8 * hl);
      bf_[j].h8[1] = *(const v8bf*)(bp + 16 + 8 * hl);
    }
#pragma unroll
    for (int i = 0; i < 4; i++)
#pragma unroll
      for (int j = 0; j < 2; j++)
        acc[i][j] = wmma_bf16(af[i].v, bf_[j].v, acc[i][j]);
    // no tail barrier: next iteration stages the other buffer.
  }

#pragma unroll
  for (int i = 0; i < 4; i++)
#pragma unroll
    for (int j = 0; j < 2; j++)
#pragma unroll
      for (int r = 0; r < 8; r++) {
        int row = m0 + wm * 64 + i * 16 + r + 8 * hl;
        int col = n0 + wn * 32 + j * 16 + mr;
        C[(size_t)row * N + col] = acc[i][j][r];
      }
}

// ---------------------------------------------------------------------------
// RMSNorm + RoPE + gate extraction.
// blockIdx.x = s; 8 waves. Lane owns d in {l, l+32, l+64, l+96} so the
// rotate-half pair (d, d+64) is intra-lane.
// ---------------------------------------------------------------------------
__global__ __launch_bounds__(256) void normrope_kernel(
    const float* __restrict__ qg, const float* __restrict__ kproj,
    const float* __restrict__ vproj,
    const float* __restrict__ cosT, const float* __restrict__ sinT,
    const float* __restrict__ qw, const float* __restrict__ kw,
    __bf16* __restrict__ q_bf, __bf16* __restrict__ k_bf,
    __bf16* __restrict__ v_bf, float* __restrict__ gate)
{
  const int s = blockIdx.x;
  const int tid = threadIdx.x;
  const int wave = tid >> 5, lane = tid & 31;
  const int d0 = lane;

  float c[4], sn[4];
#pragma unroll
  for (int e = 0; e < 4; e++) {
    c[e]  = cosT[s * DH + d0 + 32 * e];
    sn[e] = sinT[s * DH + d0 + 32 * e];
  }

  // ---- Q heads: wave w handles heads 4w..4w+3
#pragma unroll
  for (int ii = 0; ii < 4; ii++) {
    int h = wave * 4 + ii;
    const float* x = qg + (size_t)s * 8192 + h * 256;
    float v[4], ss = 0.f;
#pragma unroll
    for (int e = 0; e < 4; e++) { v[e] = x[d0 + 32 * e]; ss += v[e] * v[e]; }
#pragma unroll
    for (int m = 16; m >= 1; m >>= 1) ss += __shfl_xor(ss, m);
    float r = rsqrtf(ss * (1.f / 128.f) + 1e-6f);
#pragma unroll
    for (int e = 0; e < 4; e++) v[e] = v[e] * r * qw[d0 + 32 * e];
    float o0 = v[0] * c[0] - v[2] * sn[0];
    float o1 = v[1] * c[1] - v[3] * sn[1];
    float o2 = v[2] * c[2] + v[0] * sn[2];
    float o3 = v[3] * c[3] + v[1] * sn[3];
    __bf16* qo = q_bf + ((size_t)h * S_LEN + s) * DH;
    qo[d0] = (__bf16)o0; qo[d0 + 32] = (__bf16)o1;
    qo[d0 + 64] = (__bf16)o2; qo[d0 + 96] = (__bf16)o3;
    const float* gx = x + 128;
    float* go = gate + (size_t)s * HSZ + h * DH;
#pragma unroll
    for (int e = 0; e < 4; e++) go[d0 + 32 * e] = gx[d0 + 32 * e];
  }

  // ---- K head (wave w handles kv head w) + V conversion
  {
    int h = wave;
    const float* x = kproj + (size_t)s * (NKV * DH) + h * DH;
    float v[4], ss = 0.f;
#pragma unroll
    for (int e = 0; e < 4; e++) { v[e] = x[d0 + 32 * e]; ss += v[e] * v[e]; }
#pragma unroll
    for (int m = 16; m >= 1; m >>= 1) ss += __shfl_xor(ss, m);
    float r = rsqrtf(ss * (1.f / 128.f) + 1e-6f);
#pragma unroll
    for (int e = 0; e < 4; e++) v[e] = v[e] * r * kw[d0 + 32 * e];
    float o0 = v[0] * c[0] - v[2] * sn[0];
    float o1 = v[1] * c[1] - v[3] * sn[1];
    float o2 = v[2] * c[2] + v[0] * sn[2];
    float o3 = v[3] * c[3] + v[1] * sn[3];
    __bf16* ko = k_bf + ((size_t)h * S_LEN + s) * DH;
    ko[d0] = (__bf16)o0; ko[d0 + 32] = (__bf16)o1;
    ko[d0 + 64] = (__bf16)o2; ko[d0 + 96] = (__bf16)o3;

    const float* vx = vproj + (size_t)s * (NKV * DH) + h * DH;
    __bf16* vo = v_bf + ((size_t)h * S_LEN + s) * DH;
#pragma unroll
    for (int e = 0; e < 4; e++) vo[d0 + 32 * e] = (__bf16)vx[d0 + 32 * e];
  }
}

// ---------------------------------------------------------------------------
// Flash attention (causal, GQA 4:1) + sigmoid gating, output bf16 (S, H*D).
// Block = (128 q rows, 1 head); wave owns 16 q rows. 32-key inner blocks.
// Double-buffered: K tile prefetched with global_load_async_to_lds_b128
// (ASYNCcnt), V register-prefetched; one barrier per key block.
// ---------------------------------------------------------------------------
__global__ __launch_bounds__(256) void flash_kernel(
    const __bf16* __restrict__ q_bf, const __bf16* __restrict__ k_bf,
    const __bf16* __restrict__ v_bf, const float* __restrict__ gate,
    __bf16* __restrict__ attn_out)
{
  __shared__ __align__(16) __bf16 K_lds[2][32 * 136];   // [key][d]  == B-layout
  __shared__ __align__(16) __bf16 V_lds[2][128 * 40];   // [d][key]  == B-layout
  __shared__ __align__(16) __bf16 P_lds[8 * 16 * 32];   // per-wave P tile

  const int h   = blockIdx.y;
  const int kvh = h >> 2;                  // H/KV = 4
  const int q0  = blockIdx.x * 128;
  const int tid = threadIdx.x;
  const int wave = tid >> 5, lane = tid & 31;
  const int hl = lane >> 4, mr = lane & 15;
  const int qr = q0 + wave * 16;

  // Staging coords (2 x 16B per thread per tile)
  int srow[2], sc8[2];
#pragma unroll
  for (int l = 0; l < 2; l++) {
    int e = l * 256 + tid;                 // 0..511 octets
    srow[l] = e >> 4; sc8[l] = e & 15;
  }

  // Q fragments straight from global in WMMA A-layout (row = mr)
  Frag qfr[4];
#pragma unroll
  for (int kk = 0; kk < 4; kk++) {
    const __bf16* qp = q_bf + ((size_t)h * S_LEN + (qr + mr)) * DH + kk * 32;
    qfr[kk].h8[0] = *(const v8bf*)(qp + 8 * hl);
    qfr[kk].h8[1] = *(const v8bf*)(qp + 16 + 8 * hl);
  }

  const v8f zf = {0.f,0.f,0.f,0.f,0.f,0.f,0.f,0.f};
  float m_i[8], l_i[8];
  v8f oacc[8];
#pragma unroll
  for (int r = 0; r < 8; r++) { m_i[r] = -3.0e38f; l_i[r] = 0.f; }
#pragma unroll
  for (int dj = 0; dj < 8; dj++) oacc[dj] = zf;

  v8bf vreg[2];

  auto issue_k_async = [&](int kb, int buf) {
    const __bf16* kbase = k_bf + ((size_t)kvh * S_LEN + kb * 32) * DH;
#pragma unroll
    for (int l = 0; l < 2; l++) {
      unsigned ldsa =
          (unsigned)(unsigned long long)&K_lds[buf][srow[l] * 136 + sc8[l] * 8];
      const __bf16* g = kbase + srow[l] * DH + sc8[l] * 8;
      asm volatile("global_load_async_to_lds_b128 %0, %1, off"
                   :: "v"(ldsa), "v"(g) : "memory");
    }
  };
  auto load_v_regs = [&](int kb) {
    const __bf16* vbase = v_bf + ((size_t)kvh * S_LEN + kb * 32) * DH;
#pragma unroll
    for (int l = 0; l < 2; l++)
      vreg[l] = *(const v8bf*)(vbase + srow[l] * DH + sc8[l] * 8);
  };
  auto scatter_v = [&](int buf) {
#pragma unroll
    for (int l = 0; l < 2; l++)
#pragma unroll
      for (int q = 0; q < 8; q++)
        V_lds[buf][(sc8[l] * 8 + q) * 40 + srow[l]] = vreg[l][q];
  };

  const int nkb = (q0 + 128) >> 5;
  issue_k_async(0, 0);
  load_v_regs(0);

  for (int kb = 0; kb < nkb; kb++) {
    const int cur = kb & 1;
    scatter_v(cur);                                     // V[cur] ready (per-wave)
    asm volatile("s_wait_asynccnt 0x0" ::: "memory");   // K[cur] DMA complete
    __syncthreads();                                    // all waves: tiles ready
    if (kb + 1 < nkb) {                                 // prefetch next block
      issue_k_async(kb + 1, cur ^ 1);
      load_v_regs(kb + 1);
    }

    // ---- scores: S = Q x K^T (two 16-key column groups)
    v8f sacc[2]; sacc[0] = zf; sacc[1] = zf;
#pragma unroll
    for (int kk = 0; kk < 4; kk++)
#pragma unroll
      for (int j = 0; j < 2; j++) {
        Frag bfr;
        const __bf16* bp = &K_lds[cur][(j * 16 + mr) * 136 + kk * 32];
        bfr.h8[0] = *(const v8bf*)(bp + 8 * hl);
        bfr.h8[1] = *(const v8bf*)(bp + 16 + 8 * hl);
        sacc[j] = wmma_bf16(qfr[kk].v, bfr.v, sacc[j]);
      }

    // ---- online softmax (rows r+8*hl live in lanes of same half)
    float mnew[8];
#pragma unroll
    for (int r = 0; r < 8; r++) {
      int qrow = qr + r + 8 * hl;
      float v0 = sacc[0][r] * SOFT_SCALE;
      float v1 = sacc[1][r] * SOFT_SCALE;
      if (kb * 32 + mr      > qrow) v0 = -3.0e38f;
      if (kb * 32 + 16 + mr > qrow) v1 = -3.0e38f;
      sacc[0][r] = v0; sacc[1][r] = v1;
      float t = fmaxf(v0, v1);
#pragma unroll
      for (int m = 8; m >= 1; m >>= 1) t = fmaxf(t, __shfl_xor(t, m));
      mnew[r] = fmaxf(m_i[r], t);
    }
#pragma unroll
    for (int r = 0; r < 8; r++) {
      float p0 = __expf(sacc[0][r] - mnew[r]);
      float p1 = __expf(sacc[1][r] - mnew[r]);
      float rs = p0 + p1;
#pragma unroll
      for (int m = 8; m >= 1; m >>= 1) rs += __shfl_xor(rs, m);
      float alpha = __expf(m_i[r] - mnew[r]);
      l_i[r] = l_i[r] * alpha + rs;
      m_i[r] = mnew[r];
#pragma unroll
      for (int dj = 0; dj < 8; dj++) oacc[dj][r] = oacc[dj][r] * alpha;
      P_lds[wave * 512 + (r + 8 * hl) * 32 + mr]      = (__bf16)p0;
      P_lds[wave * 512 + (r + 8 * hl) * 32 + 16 + mr] = (__bf16)p1;
    }

    // ---- O += P x V   (LDS round-trip reshapes accumulator -> A fragment)
    Frag pf;
    const __bf16* pp = &P_lds[wave * 512 + mr * 32];
    pf.h8[0] = *(const v8bf*)(pp + 8 * hl);
    pf.h8[1] = *(const v8bf*)(pp + 16 + 8 * hl);
#pragma unroll
    for (int dj = 0; dj < 8; dj++) {
      Frag bfr;
      const __bf16* bp = &V_lds[cur][(dj * 16 + mr) * 40];
      bfr.h8[0] = *(const v8bf*)(bp + 8 * hl);
      bfr.h8[1] = *(const v8bf*)(bp + 16 + 8 * hl);
      oacc[dj] = wmma_bf16(pf.v, bfr.v, oacc[dj]);
    }
    // no tail barrier: next iteration writes the other buffers.
  }

  // ---- epilogue: normalize, gate with sigmoid, write bf16 (S, H*D)
#pragma unroll
  for (int dj = 0; dj < 8; dj++)
#pragma unroll
    for (int r = 0; r < 8; r++) {
      int row = qr + r + 8 * hl;
      int col = h * DH + dj * 16 + mr;
      float g = gate[(size_t)row * HSZ + col];
      float val = (oacc[dj][r] / l_i[r]) * (1.f / (1.f + __expf(-g)));
      attn_out[(size_t)row * HSZ + col] = (__bf16)val;
    }
}

// ---------------------------------------------------------------------------
extern "C" void kernel_launch(void* const* d_in, const int* in_sizes, int n_in,
                              void* d_out, int out_size, void* d_ws, size_t ws_size,
                              hipStream_t stream)
{
  (void)in_sizes; (void)n_in; (void)out_size; (void)ws_size;
  const float* hidden = (const float*)d_in[0];
  const float* cosT   = (const float*)d_in[1];
  const float* sinT   = (const float*)d_in[2];
  const float* Wq     = (const float*)d_in[3];
  const float* Wk     = (const float*)d_in[4];
  const float* Wv     = (const float*)d_in[5];
  const float* Wo     = (const float*)d_in[6];
  const float* qnw    = (const float*)d_in[7];
  const float* knw    = (const float*)d_in[8];

  char* ws = (char*)d_ws;
  size_t off = 0;
  auto carve = [&](size_t bytes) -> char* {
    char* p = ws + off; off += (bytes + 255) & ~(size_t)255; return p;
  };
  float*  qg    = (float*) carve((size_t)S_LEN * 8192 * 4);
  float*  kproj = (float*) carve((size_t)S_LEN * NKV * DH * 4);
  float*  vproj = (float*) carve((size_t)S_LEN * NKV * DH * 4);
  float*  gate  = (float*) carve((size_t)S_LEN * HSZ * 4);
  __bf16* q_bf  = (__bf16*)carve((size_t)NH  * S_LEN * DH * 2);
  __bf16* k_bf  = (__bf16*)carve((size_t)NKV * S_LEN * DH * 2);
  __bf16* v_bf  = (__bf16*)carve((size_t)NKV * S_LEN * DH * 2);
  __bf16* attn  = (__bf16*)carve((size_t)S_LEN * HSZ * 2);

  // 1) projections (fp32 A, fp32 weights -> bf16 tiles -> WMMA, f32 out)
  gemm_wmma_kernel<0><<<dim3(8192 / 128, S_LEN / 128), 256, 0, stream>>>(
      hidden, Wq, qg, 8192, HSZ);
  gemm_wmma_kernel<0><<<dim3(1024 / 128, S_LEN / 128), 256, 0, stream>>>(
      hidden, Wk, kproj, NKV * DH, HSZ);
  gemm_wmma_kernel<0><<<dim3(1024 / 128, S_LEN / 128), 256, 0, stream>>>(
      hidden, Wv, vproj, NKV * DH, HSZ);

  // 2) RMSNorm + RoPE + gate extraction
  normrope_kernel<<<S_LEN, 256, 0, stream>>>(qg, kproj, vproj, cosT, sinT,
                                             qnw, knw, q_bf, k_bf, v_bf, gate);

  // 3) causal flash attention + sigmoid gating (async double-buffered K/V)
  flash_kernel<<<dim3(S_LEN / 128, NH), 256, 0, stream>>>(q_bf, k_bf, v_bf,
                                                          gate, attn);

  // 4) output projection (bf16 A x fp32 Wo -> f32 out)
  gemm_wmma_kernel<1><<<dim3(HSZ / 128, S_LEN / 128), 256, 0, stream>>>(
      attn, Wo, (float*)d_out, HSZ, HSZ);
}
